// Self_Attn_12902081757223
// MI455X (gfx1250) — matrix operand order repaired and checked by reading, not compiled, not run on hardware
//
#include <hip/hip_runtime.h>

#define B_ 8
#define C_ 128
#define N_ 4096

typedef __attribute__((ext_vector_type(16))) _Float16 v16h;
typedef __attribute__((ext_vector_type(8)))  _Float16 v8h;
typedef __attribute__((ext_vector_type(8)))  float    v8f;

// Build a 16-half WMMA fragment from two 16-byte chunks (global or LDS).
static __device__ __forceinline__ v16h ld16(const _Float16* p0, const _Float16* p1) {
  v8h a = *(const v8h*)p0;
  v8h b = *(const v8h*)p1;
  v16h r;
#pragma unroll
  for (int i = 0; i < 8; ++i) { r[i] = a[i]; r[i + 8] = b[i]; }
  return r;
}

// CDNA5 async copy: 16B per lane, global -> LDS, bypassing VGPRs.
// Tracked with ASYNCcnt (in-order completion per wave).
static __device__ __forceinline__ void async_copy16(const void* gptr, void* lptr) {
  // Generic LDS pointer: low 32 bits are the wave-relative LDS byte offset.
  unsigned lds = (unsigned)(uintptr_t)lptr;
  asm volatile("global_load_async_to_lds_b128 %0, %1, off"
               :: "v"(lds), "v"(gptr) : "memory");
}
#define WAIT_ASYNCCNT(n) asm volatile("s_wait_asynccnt " #n ::: "memory")

// ---------------------------------------------------------------------------
// Kernel 1: 1x1-conv projections.  f = Wf x + bf, g = Wg x + bg.
// Outputs fT,gT as f16 [B][N][C] (row-major, C contiguous) and x16 as f16
// [B][C][N] — layouts chosen so WMMA fragment loads are contiguous.
// One block = 16 pixels, 128 threads (thread = output channel).
// ---------------------------------------------------------------------------
__global__ __launch_bounds__(128) void proj_kernel(
    const float* __restrict__ x, const float* __restrict__ Wf,
    const float* __restrict__ bf, const float* __restrict__ Wg,
    const float* __restrict__ bg,
    _Float16* __restrict__ fT, _Float16* __restrict__ gT,
    _Float16* __restrict__ x16) {
  __shared__ __align__(16) float sX[C_][16];   // [c][pixel]
  const int t = threadIdx.x;
  const int pix0 = blockIdx.x * 16;
  const int b = pix0 >> 12;           // / 4096
  const int n0 = pix0 & (N_ - 1);

  const size_t xoff = ((size_t)(b * C_ + t)) * N_ + n0;
  const float4* xv = (const float4*)(x + xoff);
#pragma unroll
  for (int q = 0; q < 4; ++q) {
    float4 v = xv[q];
    sX[t][q * 4 + 0] = v.x; sX[t][q * 4 + 1] = v.y;
    sX[t][q * 4 + 2] = v.z; sX[t][q * 4 + 3] = v.w;
  }
  // f16 copy of x in [C][N] layout (contiguous 32B store)
  {
    v8h h0, h1;
#pragma unroll
    for (int i = 0; i < 8; ++i) {
      h0[i] = (_Float16)sX[t][i];
      h1[i] = (_Float16)sX[t][8 + i];
    }
    v8h* dst = (v8h*)(x16 + xoff);
    dst[0] = h0; dst[1] = h1;
  }
  __syncthreads();

  float accf[16], accg[16];
  const float bfv = bf[t], bgv = bg[t];
#pragma unroll
  for (int p = 0; p < 16; ++p) { accf[p] = bfv; accg[p] = bgv; }
  const float* wfrow = Wf + t * C_;   // f[o,n] = sum_c Wf[o,c] x[c,n]
  const float* wgrow = Wg + t * C_;
  for (int k = 0; k < C_; ++k) {
    const float wf = wfrow[k], wg = wgrow[k];
#pragma unroll
    for (int p = 0; p < 16; ++p) {
      const float xx = sX[k][p];
      accf[p] = fmaf(wf, xx, accf[p]);
      accg[p] = fmaf(wg, xx, accg[p]);
    }
  }
#pragma unroll
  for (int p = 0; p < 16; ++p) {
    const size_t o = ((size_t)(b * N_ + n0 + p)) * C_ + t;  // [B][N][C]
    fT[o] = (_Float16)accf[p];
    gT[o] = (_Float16)accg[p];
  }
}

// ---------------------------------------------------------------------------
// Kernel 2: fused flash attention, double-buffered async-LDS key tiles.
//   S = f^T g   (16 queries x 64 keys per wave per step, K=C=128)
//   online softmax over keys
//   O += P * x^T  (accumulate 16 x 128 output channels)
//   out = gamma * O / l + x
// Block: 256 threads = 8 waves; wave w owns query rows n0..n0+15.
// Key tiles (g^T 16KB + x16 16KB) fetched with global_load_async_to_lds_b128
// into buffer 1-p while buffer p is consumed (ASYNCcnt pipelining).
// ---------------------------------------------------------------------------
__global__ __launch_bounds__(256) void attn_kernel(
    const float* __restrict__ xin, const float* __restrict__ gamma,
    const _Float16* __restrict__ fT, const _Float16* __restrict__ gT,
    const _Float16* __restrict__ x16, float* __restrict__ out) {
  __shared__ __align__(16) _Float16 sG[2][64][C_];    // g^T tiles  [m][c] 2x16KB
  __shared__ __align__(16) _Float16 sXk[2][C_][64];   // x16 tiles  [c][m] 2x16KB
  __shared__ __align__(16) _Float16 sP[8][16][64];    // per-wave P 16KB

  const int tid  = threadIdx.x;
  const int w    = tid >> 5;
  const int lane = tid & 31;
  const int r    = lane & 15;
  const int half = lane >> 4;
  const int b    = blockIdx.y;
  const int n0   = blockIdx.x * 128 + w * 16;

  const float gam = gamma[0];
  const _Float16* gbase = gT + (size_t)b * N_ * C_;
  const _Float16* xbase = x16 + (size_t)b * C_ * N_;

  // Stage one 64-key tile (32KB) into LDS buffer `buf`:
  // 2048 x 16B chunks, 8 async instructions per thread.
  auto stage_tile = [&](int m0, int buf) {
#pragma unroll
    for (int i = 0; i < 4; ++i) {                 // g^T: contiguous 16KB
      const int idx = tid + i * 256;              // 0..1023
      async_copy16(gbase + (size_t)m0 * C_ + idx * 8, &sG[buf][0][0] + idx * 8);
    }
#pragma unroll
    for (int i = 0; i < 4; ++i) {                 // x16: 128 rows x 128B
      const int idx = tid + i * 256;              // 0..1023
      const int row = idx >> 3, col = idx & 7;
      async_copy16(xbase + (size_t)row * N_ + m0 + col * 8,
                   &sXk[buf][0][0] + idx * 8);
    }
  };

  // Loop-invariant A fragments: f^T rows n0..n0+15, all 128 k (4 chunks of 32).
  // A layout (16-bit 16x32): lane<16 -> row r, k = [k0..k0+8) & [k0+16..k0+24);
  //                          lane>=16 -> row r, k = [k0+8..) & [k0+24..).
  v16h aF[4];
  {
    const _Float16* frow = fT + ((size_t)(b * N_ + n0 + r)) * C_;
#pragma unroll
    for (int kc = 0; kc < 4; ++kc) {
      const _Float16* pa = frow + kc * 32 + half * 8;
      aF[kc] = ld16(pa, pa + 16);
    }
  }

  v8f o[8];
#pragma unroll
  for (int ci = 0; ci < 8; ++ci) o[ci] = (v8f){};
  float rmax[8], rsum[8];
#pragma unroll
  for (int j = 0; j < 8; ++j) { rmax[j] = -3.0e38f; rsum[j] = 0.0f; }

  constexpr int NT = N_ / 64;   // 64 key tiles
  int p = 0;
  stage_tile(0, 0);             // prologue: tile 0 in flight

  for (int t = 0; t < NT; ++t) {
    const int m0 = t * 64;
    if (t + 1 < NT) {
      stage_tile(m0 + 64, 1 - p);   // overlap next fetch with this compute
      WAIT_ASYNCCNT(0x8);           // in-order: tile t's 8 ops have drained
    } else {
      WAIT_ASYNCCNT(0x0);
    }
    __syncthreads();                // tile t visible to all waves

    // ---- S = f^T g : 4 key sub-tiles x 4 K-chunks = 16 WMMA ----
    // B layout (16-bit 32x16): lane<16 -> col r, k=[k0..k0+8) & [k0+16..k0+24);
    //                          lane>=16 -> col r, k=[k0+8..) & [k0+24..).
    v8f st[4];
#pragma unroll
    for (int mt = 0; mt < 4; ++mt) {
      v8f s = {};
#pragma unroll
      for (int kc = 0; kc < 4; ++kc) {
        const _Float16* pb = &sG[p][mt * 16 + r][kc * 32 + half * 8];
        v16h bG = ld16(pb, pb + 16);
        s = __builtin_amdgcn_wmma_f32_16x16x32_f16(false, aF[kc], false, bG,
                                                   (short)0, s, false, false);
      }
      st[mt] = s;
    }

    // ---- online softmax over this key tile ----
    // D layout: lane holds rows j+8*half (j = vgpr idx), col = r.
    float cf[8], ps[8];
#pragma unroll
    for (int j = 0; j < 8; ++j) {
      float v = fmaxf(fmaxf(st[0][j], st[1][j]), fmaxf(st[2][j], st[3][j]));
#pragma unroll
      for (int msk = 1; msk < 16; msk <<= 1)
        v = fmaxf(v, __shfl_xor(v, msk, 32));     // reduce within 16-lane group
      const float mn = fmaxf(rmax[j], v);
      cf[j] = __expf(rmax[j] - mn);
      rmax[j] = mn;
      ps[j] = 0.0f;
    }
#pragma unroll
    for (int ci = 0; ci < 8; ++ci)
#pragma unroll
      for (int j = 0; j < 8; ++j) o[ci][j] *= cf[j];
#pragma unroll
    for (int mt = 0; mt < 4; ++mt)
#pragma unroll
      for (int j = 0; j < 8; ++j) {
        const float e = __expf(st[mt][j] - rmax[j]);
        st[mt][j] = e;
        ps[j] += e;
      }
#pragma unroll
    for (int j = 0; j < 8; ++j) {
      float v = ps[j];
#pragma unroll
      for (int msk = 1; msk < 16; msk <<= 1) v += __shfl_xor(v, msk, 32);
      rsum[j] = rsum[j] * cf[j] + v;
    }

    // ---- P (D layout) -> per-wave LDS row-major [16 n][64 m] in f16 ----
#pragma unroll
    for (int mt = 0; mt < 4; ++mt)
#pragma unroll
      for (int j = 0; j < 8; ++j)
        sP[w][j + 8 * half][mt * 16 + r] = (_Float16)st[mt][j];
    // same-wave DS ops are in-order; loads below see these stores

    // ---- O += P * x^T : 2 K-chunks x 8 channel tiles = 16 WMMA ----
#pragma unroll
    for (int kc = 0; kc < 2; ++kc) {
      const _Float16* pr = &sP[w][r][kc * 32 + half * 8];
      v16h aP = ld16(pr, pr + 16);
#pragma unroll
      for (int ci = 0; ci < 8; ++ci) {
        const _Float16* pb = &sXk[p][ci * 16 + r][kc * 32 + half * 8];
        v16h bX = ld16(pb, pb + 16);
        o[ci] = __builtin_amdgcn_wmma_f32_16x16x32_f16(false, aP, false, bX,
                                                       (short)0, o[ci], false,
                                                       false);
      }
    }

    __syncthreads();   // buffer p free before iter t+1 refills it
    p ^= 1;
  }

  // ---- epilogue: out = gamma * O / l + x ----
#pragma unroll
  for (int ci = 0; ci < 8; ++ci) {
#pragma unroll
    for (int j = 0; j < 8; ++j) {
      const size_t idx =
          ((size_t)(b * C_ + ci * 16 + r)) * N_ + n0 + j + 8 * half;
      out[idx] = gam * (o[ci][j] / rsum[j]) + xin[idx];
    }
  }
}

extern "C" void kernel_launch(void* const* d_in, const int* in_sizes, int n_in,
                              void* d_out, int out_size, void* d_ws,
                              size_t ws_size, hipStream_t stream) {
  (void)in_sizes; (void)n_in; (void)out_size; (void)ws_size;
  const float* x     = (const float*)d_in[0];
  const float* Wf    = (const float*)d_in[1];
  const float* bf    = (const float*)d_in[2];
  const float* Wg    = (const float*)d_in[3];
  const float* bg    = (const float*)d_in[4];
  const float* gamma = (const float*)d_in[5];
  float* outp = (float*)d_out;

  const size_t elems = (size_t)B_ * N_ * C_;  // 4M f16 each
  _Float16* fT  = (_Float16*)d_ws;
  _Float16* gT  = fT + elems;
  _Float16* x16 = gT + elems;

  proj_kernel<<<(B_ * N_) / 16, 128, 0, stream>>>(x, Wf, bf, Wg, bg, fT, gT,
                                                  x16);
  attn_kernel<<<dim3(N_ / 128, B_), 256, 0, stream>>>(x, gamma, fT, gT, x16,
                                                      outp);
}
